// TransformerBlock_52939766890628
// MI455X (gfx1250) — matrix operand order repaired and checked by reading, not compiled, not run on hardware
//
#include <hip/hip_runtime.h>
#include <hip/hip_bf16.h>

// ---------------- constants for this problem ----------------
#define BATCH 2
#define SLEN  2048
#define DMODEL 2048
#define NHEAD 32
#define DHEAD 64
#define FDIM  8192
#define MROWS (BATCH * SLEN)          // 4096

typedef __attribute__((ext_vector_type(16))) __bf16 v16bf;
typedef __attribute__((ext_vector_type(8)))  float  v8f;

union Frag {
    v16bf v;
    uint4 u4[2];
};

static __device__ __forceinline__ v8f wmma_bf16(v16bf a, v16bf b, v8f c) {
    return __builtin_amdgcn_wmma_f32_16x16x32_bf16(
        /*neg_a=*/false, a, /*neg_b=*/false, b,
        /*c_mod=*/(short)0, c, /*reuse_a=*/false, /*reuse_b=*/false);
}

static __device__ __forceinline__ v8f vzero8() {
    v8f v;
#pragma unroll
    for (int i = 0; i < 8; ++i) v[i] = 0.0f;
    return v;
}

// Async DMA: global -> LDS, 16 bytes per lane, tracked by ASYNCcnt.
// lds = LDS byte address (low 32 bits of the generic pointer), g = global addr.
static __device__ __forceinline__ void async_copy16(unsigned lds, const void* g) {
    asm volatile("global_load_async_to_lds_b128 %0, %1, off"
                 :
                 : "v"(lds), "v"(g)
                 : "memory");
}
static __device__ __forceinline__ unsigned lds_addr(const void* p) {
    return (unsigned)(unsigned long long)p;   // low 32 bits = LDS offset
}

// ---------------- weight convert + transpose: W[R,C] f32 -> Wt[C,R] bf16 ----
__global__ void __launch_bounds__(256)
conv_transpose_kernel(const float* __restrict__ in, __bf16* __restrict__ out,
                      int R, int C) {
    __shared__ float tile[32][33];
    int c0 = blockIdx.x * 32;
    int r0 = blockIdx.y * 32;
#pragma unroll
    for (int j = 0; j < 4; ++j) {
        int r = r0 + threadIdx.y + j * 8;
        tile[threadIdx.y + j * 8][threadIdx.x] = in[(long)r * C + c0 + threadIdx.x];
    }
    __syncthreads();
#pragma unroll
    for (int j = 0; j < 4; ++j) {
        int c = c0 + threadIdx.y + j * 8;
        out[(long)c * R + r0 + threadIdx.x] = (__bf16)tile[threadIdx.x][threadIdx.y + j * 8];
    }
}

// ---------------- RMSNorm: f32 row -> bf16 row -----------------------------
__global__ void __launch_bounds__(256)
rmsnorm_kernel(const float* __restrict__ x, const float* __restrict__ w,
               __bf16* __restrict__ out) {
    long base = (long)blockIdx.x * DMODEL;
    float ss = 0.0f;
    for (int i = threadIdx.x; i < DMODEL; i += 256) {
        float t = x[base + i];
        ss += t * t;
    }
    __shared__ float red[256];
    red[threadIdx.x] = ss;
    __syncthreads();
    for (int st = 128; st > 0; st >>= 1) {
        if (threadIdx.x < st) red[threadIdx.x] += red[threadIdx.x + st];
        __syncthreads();
    }
    float rr = rsqrtf(red[0] * (1.0f / DMODEL) + 1e-6f);
    for (int i = threadIdx.x; i < DMODEL; i += 256)
        out[base + i] = (__bf16)(x[base + i] * rr * w[i]);
}

// ---------------- RoPE on bf16 [B*S, D], interleaved pairs -----------------
__global__ void __launch_bounds__(256)
rope_kernel(__bf16* __restrict__ v, const float* __restrict__ cs,
            const float* __restrict__ sn, float scale) {
    long i = (long)blockIdx.x * blockDim.x + threadIdx.x;   // pair index
    const long total = (long)MROWS * NHEAD * (DHEAD / 2);
    if (i >= total) return;
    int dh2 = (int)(i & 31);
    int hh  = (int)((i >> 5) & (NHEAD - 1));
    long bs = i >> 10;                 // 0..MROWS-1
    int s   = (int)(bs & (SLEN - 1));
    long base = bs * DMODEL + hh * DHEAD + dh2 * 2;
    float c  = cs[s * 32 + dh2];
    float sv = sn[s * 32 + dh2];
    float v0 = (float)v[base];
    float v1 = (float)v[base + 1];
    v[base]     = (__bf16)((v0 * c - v1 * sv) * scale);
    v[base + 1] = (__bf16)((v0 * sv + v1 * c) * scale);
}

// ---------------- silu(gate) * up, bf16, writes over gate ------------------
__global__ void __launch_bounds__(256)
silu_mul_kernel(__bf16* __restrict__ gate, const __bf16* __restrict__ up) {
    long i = (long)blockIdx.x * blockDim.x + threadIdx.x;
    if (i >= (long)MROWS * FDIM) return;
    float g = (float)gate[i];
    float u = (float)up[i];
    float s = g / (1.0f + __expf(-g));
    gate[i] = (__bf16)(s * u);
}

// ---------------- tiled WMMA GEMM: C[M,N] = A[M,K] @ Bt[N,K]^T -------------
// block: 256 threads (8 waves), tile 128x128, K-step 64, double-buffered LDS
// filled by GLOBAL_LOAD_ASYNC_TO_LDS_B128 (ASYNCcnt) to overlap DMA with WMMA.
enum { OUT_BF16 = 0, OUT_VT = 1, OUT_F32_RES = 2 };

template <int MODE>
__global__ void __launch_bounds__(256)
gemm_bf16(const __bf16* __restrict__ A, const __bf16* __restrict__ Bt,
          void* __restrict__ outp, const float* __restrict__ resid,
          int K, int lda, int ldb, int ldc) {
    __shared__ __align__(16) __bf16 sA[2][128 * 72];  // 64 K elems + 8 pad
    __shared__ __align__(16) __bf16 sB[2][128 * 72];

    const int tid  = threadIdx.x;
    const int n0   = blockIdx.x * 128;
    const int m0   = blockIdx.y * 128;
    const int wid  = tid >> 5;
    const int lane = tid & 31;
    const int hl   = lane >> 4;
    const int l15  = lane & 15;
    const int wm   = (wid >> 2) * 64;   // 0,64
    const int wn   = (wid & 3) * 32;    // 0,32,64,96

    // per-thread staging slots (4 x 16B for A, same for B)
    const int srow = tid >> 1;              // unused helper removed below
    (void)srow;

    auto issue_tile = [&](int kb, int buf) {
#pragma unroll
        for (int i = 0; i < 4; ++i) {
            int idx = tid + i * 256;
            int row = idx >> 3, c8 = idx & 7;
            async_copy16(lds_addr(&sA[buf][row * 72 + c8 * 8]),
                         A + (long)(m0 + row) * lda + kb + c8 * 8);
            async_copy16(lds_addr(&sB[buf][row * 72 + c8 * 8]),
                         Bt + (long)(n0 + row) * ldb + kb + c8 * 8);
        }
    };

    v8f acc[4][2];
#pragma unroll
    for (int fm = 0; fm < 4; ++fm)
#pragma unroll
        for (int fn = 0; fn < 2; ++fn) acc[fm][fn] = vzero8();

    issue_tile(0, 0);

    for (int kb = 0; kb < K; kb += 64) {
        const int cur = (kb >> 6) & 1;
        if (kb + 64 < K) {
            issue_tile(kb + 64, cur ^ 1);
            // 8 ops of the next tile may stay in flight; current tile must land
            asm volatile("s_wait_asynccnt 0x8" ::: "memory");
        } else {
            asm volatile("s_wait_asynccnt 0x0" ::: "memory");
        }
        __syncthreads();

#pragma unroll
        for (int ks = 0; ks < 2; ++ks) {
            Frag af[4], bq[2];
#pragma unroll
            for (int fm = 0; fm < 4; ++fm) {
                const __bf16* p = &sA[cur][(wm + fm * 16 + l15) * 72 + ks * 32];
                af[fm].u4[0] = *(const uint4*)(p + hl * 8);
                af[fm].u4[1] = *(const uint4*)(p + 16 + hl * 8);
            }
#pragma unroll
            for (int fn = 0; fn < 2; ++fn) {
                const __bf16* p = &sB[cur][(wn + fn * 16 + l15) * 72 + ks * 32];
                bq[fn].u4[0] = *(const uint4*)(p + hl * 8);
                bq[fn].u4[1] = *(const uint4*)(p + 16 + hl * 8);
            }
#pragma unroll
            for (int fm = 0; fm < 4; ++fm)
#pragma unroll
                for (int fn = 0; fn < 2; ++fn)
                    acc[fm][fn] = wmma_bf16(af[fm].v, bq[fn].v, acc[fm][fn]);
        }
        __syncthreads();   // all waves done reading `cur` before it is refilled
    }

#pragma unroll
    for (int fm = 0; fm < 4; ++fm)
#pragma unroll
        for (int fn = 0; fn < 2; ++fn)
#pragma unroll
            for (int r = 0; r < 8; ++r) {
                int m = m0 + wm + fm * 16 + r + 8 * hl;
                int n = n0 + wn + fn * 16 + l15;
                float val = acc[fm][fn][r];
                if (MODE == OUT_BF16) {
                    ((__bf16*)outp)[(long)m * ldc + n] = (__bf16)val;
                } else if (MODE == OUT_VT) {
                    // m = b*SLEN+s ; n = h*DHEAD+d ; write VT[b][h][d][s]
                    int bb = m >> 11, s = m & (SLEN - 1);
                    int hh = n >> 6, d = n & (DHEAD - 1);
                    ((__bf16*)outp)[(((long)(bb * NHEAD + hh) * DHEAD + d) << 11) + s] =
                        (__bf16)val;
                } else {
                    long idx = (long)m * ldc + n;
                    ((float*)outp)[idx] = val + resid[idx];
                }
            }
}

// ---------------- flash attention: 1 wave per 16-row Q block ---------------
// q pre-scaled by 1/sqrt(Dh). k: [B,S,H,Dh] bf16. vt: [B,H,Dh,S] bf16.
__global__ void __launch_bounds__(128)
flash_attn_kernel(const __bf16* __restrict__ q, const __bf16* __restrict__ k,
                  const __bf16* __restrict__ vt, __bf16* __restrict__ out) {
    __shared__ __align__(16) __bf16 pls[4][16][136];

    const int tid  = threadIdx.x;
    const int w    = tid >> 5;
    const int lane = tid & 31;
    const int hl   = lane >> 4;
    const int l15  = lane & 15;

    const int gw = blockIdx.x * 4 + w;
    const int qb = gw & (SLEN / 16 - 1);
    const int hh = (gw >> 7) & (NHEAD - 1);
    const int b  = gw >> 12;
    const int q0 = qb * 16;

    // Q fragments (A layout), Dh=64 -> 2 k-steps
    Frag qa[2];
    {
        const __bf16* qrow = q + ((long)(b * SLEN + q0 + l15) * DMODEL + hh * DHEAD);
#pragma unroll
        for (int kk = 0; kk < 2; ++kk) {
            qa[kk].u4[0] = *(const uint4*)(qrow + kk * 32 + hl * 8);
            qa[kk].u4[1] = *(const uint4*)(qrow + kk * 32 + 16 + hl * 8);
        }
    }

    v8f oa[4];
#pragma unroll
    for (int f = 0; f < 4; ++f) oa[f] = vzero8();
    float mi[8], li[8];
#pragma unroll
    for (int r = 0; r < 8; ++r) { mi[r] = -1e30f; li[r] = 0.0f; }

    for (int k0 = 0; k0 <= q0; k0 += 128) {
        // prefetch next K/V tile into the cache hierarchy (global_prefetch_b8)
        if (k0 + 128 <= q0) {
            __builtin_prefetch(
                k + ((long)(b * SLEN + k0 + 128 + l15 * 8) * DMODEL + hh * DHEAD), 0, 3);
            __builtin_prefetch(
                vt + ((long)((b * NHEAD + hh) * DHEAD + lane * 2) * SLEN + k0 + 128), 0, 3);
        }

        // ---- scores S (16 x 128), 8 accumulators
        v8f sa[8];
#pragma unroll
        for (int j = 0; j < 8; ++j) sa[j] = vzero8();
#pragma unroll
        for (int j = 0; j < 8; ++j) {
            const __bf16* krow =
                k + ((long)(b * SLEN + k0 + j * 16 + l15) * DMODEL + hh * DHEAD);
#pragma unroll
            for (int kk = 0; kk < 2; ++kk) {
                Frag kb;
                kb.u4[0] = *(const uint4*)(krow + kk * 32 + hl * 8);
                kb.u4[1] = *(const uint4*)(krow + kk * 32 + 16 + hl * 8);
                sa[j] = wmma_bf16(qa[kk].v, kb.v, sa[j]);
            }
        }

        // ---- causal mask on the diagonal tile
        if (q0 - k0 < 128) {
#pragma unroll
            for (int j = 0; j < 8; ++j)
#pragma unroll
                for (int r = 0; r < 8; ++r) {
                    int col = k0 + j * 16 + l15;
                    int row = q0 + r + 8 * hl;
                    if (col > row) sa[j][r] = -1e9f;
                }
        }

        // ---- online softmax per row (rows live in 16-lane halves)
#pragma unroll
        for (int r = 0; r < 8; ++r) {
            float mx = sa[0][r];
#pragma unroll
            for (int j = 1; j < 8; ++j) mx = fmaxf(mx, sa[j][r]);
#pragma unroll
            for (int off = 8; off >= 1; off >>= 1)
                mx = fmaxf(mx, __shfl_xor(mx, off, 32));
            float mnew  = fmaxf(mi[r], mx);
            float alpha = __expf(mi[r] - mnew);
            float rs = 0.0f;
#pragma unroll
            for (int j = 0; j < 8; ++j) {
                float p = __expf(sa[j][r] - mnew);
                rs += p;
                pls[w][r + 8 * hl][j * 16 + l15] = (__bf16)p;
            }
#pragma unroll
            for (int off = 8; off >= 1; off >>= 1)
                rs += __shfl_xor(rs, off, 32);
            li[r] = li[r] * alpha + rs;
            mi[r] = mnew;
#pragma unroll
            for (int f = 0; f < 4; ++f) oa[f][r] *= alpha;
        }

        __builtin_amdgcn_wave_barrier();

        // ---- O += P (16x128) @ V (128x64): P from LDS as A-frags, V^T rows as B
#pragma unroll
        for (int kk = 0; kk < 4; ++kk) {
            Frag pa;
            const __bf16* prow = &pls[w][l15][kk * 32];
            pa.u4[0] = *(const uint4*)(prow + hl * 8);
            pa.u4[1] = *(const uint4*)(prow + 16 + hl * 8);
#pragma unroll
            for (int f = 0; f < 4; ++f) {
                const __bf16* vrow =
                    vt + ((long)((b * NHEAD + hh) * DHEAD + f * 16 + l15) * SLEN +
                          k0 + kk * 32);
                Frag vb;
                vb.u4[0] = *(const uint4*)(vrow + hl * 8);
                vb.u4[1] = *(const uint4*)(vrow + 16 + hl * 8);
                oa[f] = wmma_bf16(pa.v, vb.v, oa[f]);
            }
        }
        __builtin_amdgcn_wave_barrier();
    }

    // ---- epilogue: O / l, bf16 out [B*S, D]
#pragma unroll
    for (int r = 0; r < 8; ++r) {
        float inv = 1.0f / li[r];
        long m = (long)(b * SLEN + q0 + r + 8 * hl);
#pragma unroll
        for (int f = 0; f < 4; ++f)
            out[m * DMODEL + hh * DHEAD + f * 16 + l15] = (__bf16)(oa[f][r] * inv);
    }
}

// ---------------- orchestration --------------------------------------------
extern "C" void kernel_launch(void* const* d_in, const int* in_sizes, int n_in,
                              void* d_out, int out_size, void* d_ws, size_t ws_size,
                              hipStream_t stream) {
    const float* x    = (const float*)d_in[0];
    const float* fcos = (const float*)d_in[1];
    const float* fsin = (const float*)d_in[2];
    // d_in[3] = mask (bool) — causal structure is known, unused
    const float* wq = (const float*)d_in[4];
    const float* wk = (const float*)d_in[5];
    const float* wv = (const float*)d_in[6];
    const float* wo = (const float*)d_in[7];
    const float* wg = (const float*)d_in[8];
    const float* wu = (const float*)d_in[9];
    const float* wd = (const float*)d_in[10];
    const float* n1 = (const float*)d_in[11];
    const float* n2 = (const float*)d_in[12];
    float* out = (float*)d_out;

    const long MD = (long)MROWS * DMODEL;   // 8.39M
    const long MF = (long)MROWS * FDIM;     // 33.6M
    const long DD = (long)DMODEL * DMODEL;  // 4.19M
    const long DF = (long)DMODEL * FDIM;    // 16.8M

    char* ws = (char*)d_ws;
    long off = 0;
    __bf16* hn   = (__bf16*)(ws + off); off += MD * 2;
    __bf16* qb   = (__bf16*)(ws + off); off += MD * 2;
    __bf16* kb   = (__bf16*)(ws + off); off += MD * 2;
    __bf16* vtb  = (__bf16*)(ws + off); off += MD * 2;
    __bf16* attn = (__bf16*)(ws + off); off += MD * 2;
    float*  hres = (float*) (ws + off); off += MD * 4;
    __bf16* hn2  = (__bf16*)(ws + off); off += MD * 2;
    __bf16* gate = (__bf16*)(ws + off); off += MF * 2;
    __bf16* up   = (__bf16*)(ws + off); off += MF * 2;
    __bf16* wqT  = (__bf16*)(ws + off); off += DD * 2;
    __bf16* wkT  = (__bf16*)(ws + off); off += DD * 2;
    __bf16* wvT  = (__bf16*)(ws + off); off += DD * 2;
    __bf16* woT  = (__bf16*)(ws + off); off += DD * 2;
    __bf16* wgT  = (__bf16*)(ws + off); off += DF * 2;
    __bf16* wuT  = (__bf16*)(ws + off); off += DF * 2;
    __bf16* wdT  = (__bf16*)(ws + off); off += DF * 2;

    dim3 tb32(32, 8);
    // weight convert + transpose: in [K,N] -> out [N,K]
    conv_transpose_kernel<<<dim3(DMODEL/32, DMODEL/32), tb32, 0, stream>>>(wq, wqT, DMODEL, DMODEL);
    conv_transpose_kernel<<<dim3(DMODEL/32, DMODEL/32), tb32, 0, stream>>>(wk, wkT, DMODEL, DMODEL);
    conv_transpose_kernel<<<dim3(DMODEL/32, DMODEL/32), tb32, 0, stream>>>(wv, wvT, DMODEL, DMODEL);
    conv_transpose_kernel<<<dim3(DMODEL/32, DMODEL/32), tb32, 0, stream>>>(wo, woT, DMODEL, DMODEL);
    conv_transpose_kernel<<<dim3(FDIM/32,   DMODEL/32), tb32, 0, stream>>>(wg, wgT, DMODEL, FDIM);
    conv_transpose_kernel<<<dim3(FDIM/32,   DMODEL/32), tb32, 0, stream>>>(wu, wuT, DMODEL, FDIM);
    conv_transpose_kernel<<<dim3(DMODEL/32, FDIM/32),   tb32, 0, stream>>>(wd, wdT, FDIM, DMODEL);

    // hn = rmsnorm(x) in bf16
    rmsnorm_kernel<<<MROWS, 256, 0, stream>>>(x, n1, hn);

    // Q,K,V projections
    dim3 gqkv(DMODEL / 128, MROWS / 128);
    gemm_bf16<OUT_BF16><<<gqkv, 256, 0, stream>>>(hn, wqT, qb, nullptr, DMODEL, DMODEL, DMODEL, DMODEL);
    gemm_bf16<OUT_BF16><<<gqkv, 256, 0, stream>>>(hn, wkT, kb, nullptr, DMODEL, DMODEL, DMODEL, DMODEL);
    gemm_bf16<OUT_VT ><<<gqkv, 256, 0, stream>>>(hn, wvT, vtb, nullptr, DMODEL, DMODEL, DMODEL, DMODEL);

    // RoPE (q folded with 1/sqrt(Dh))
    const long pairs = (long)MROWS * NHEAD * (DHEAD / 2);
    rope_kernel<<<(unsigned)(pairs / 256), 256, 0, stream>>>(qb, fcos, fsin, 0.125f);
    rope_kernel<<<(unsigned)(pairs / 256), 256, 0, stream>>>(kb, fcos, fsin, 1.0f);

    // flash attention: B*H*(S/16) waves, 4 waves per block
    flash_attn_kernel<<<(BATCH * NHEAD * (SLEN / 16)) / 4, 128, 0, stream>>>(qb, kb, vtb, attn);

    // h = x + attn @ wo
    gemm_bf16<OUT_F32_RES><<<gqkv, 256, 0, stream>>>(attn, woT, hres, x, DMODEL, DMODEL, DMODEL, DMODEL);

    // hn2 = rmsnorm(h)
    rmsnorm_kernel<<<MROWS, 256, 0, stream>>>(hres, n2, hn2);

    // FFN
    dim3 gff(FDIM / 128, MROWS / 128);
    gemm_bf16<OUT_BF16><<<gff, 256, 0, stream>>>(hn2, wgT, gate, nullptr, DMODEL, DMODEL, DMODEL, FDIM);
    gemm_bf16<OUT_BF16><<<gff, 256, 0, stream>>>(hn2, wuT, up,   nullptr, DMODEL, DMODEL, DMODEL, FDIM);
    silu_mul_kernel<<<(unsigned)(MF / 256), 256, 0, stream>>>(gate, up);

    // out = h + act @ wd
    dim3 gdown(DMODEL / 128, MROWS / 128);
    gemm_bf16<OUT_F32_RES><<<gdown, 256, 0, stream>>>(gate, wdT, out, hres, FDIM, FDIM, FDIM, DMODEL);
}
